// MoEModel_87737591922766
// MI455X (gfx1250) — compile-verified, matrix operand order checked by reading, not run on hardware
//
#include <hip/hip_runtime.h>

// ---------------- types ----------------
typedef __bf16 bf16_t;
typedef __attribute__((ext_vector_type(16))) __bf16 v16bf;
typedef __attribute__((ext_vector_type(8)))  __bf16 v8bf;
typedef __attribute__((ext_vector_type(4)))  __bf16 v4bf;
typedef __attribute__((ext_vector_type(8)))  float  v8f;

#define B_TOT  65536
#define DIM    1024      // D
#define INDIM  2048      // IN = 2*D
#define NEXP   4         // E
#define OUTD   128       // OUT
#define BM     32        // batch rows per block (two WMMA M tiles)
#define NTHR   512       // 16 waves
#define EPITCH 2056      // INDIM + 8 bf16 (row pitch, bank-conflict pad, 16B-aligned rows)
#define HPITCH 1032      // DIM + 8 bf16

union V16U { v16bf v; v8bf h[2]; };
union BFPK { __bf16 b[2]; unsigned int u; };

__device__ __forceinline__ v8f wmma_bf16(v16bf a, v16bf b, v8f c) {
#if defined(__HIP_DEVICE_COMPILE__)
  return __builtin_amdgcn_wmma_f32_16x16x32_bf16(false, a, false, b, (short)0, c, false, false);
#else
  return c;
#endif
}

__device__ __forceinline__ v16bf ldB(const bf16_t* p) {
  // 16 contiguous bf16 per lane (fragment-ordered weights): two b128 loads
  V16U u;
  u.h[0] = *(const v8bf*)(p);
  u.h[1] = *(const v8bf*)(p + 8);
  return u.v;
}

// ---- pre-pass: f32 weights -> bf16 in WMMA B-fragment tile order ----
// B-matrix 32x16 bf16 layout: lane = n + 16*hi ; slot s -> kk = (s&7) + 8*hi + 16*(s>>3)
// tile (e, nT, kS) stored as 32 lanes x 16 bf16 contiguous (1024 B).

__global__ void pack_w1(const float* __restrict__ W1, bf16_t* __restrict__ W1b) {
  unsigned g = blockIdx.x * 256u + threadIdx.x;       // DWORD index, total 4*64*64*256
  unsigned tile = g >> 8, d = g & 255u;
  unsigned lane = d >> 3, sp = d & 7u;
  unsigned hi = lane >> 4, n = lane & 15u;
  unsigned e = tile >> 12, nT = (tile >> 6) & 63u, kS = tile & 63u;
  BFPK pk;
#pragma unroll
  for (int j = 0; j < 2; ++j) {
    unsigned s  = sp * 2u + (unsigned)j;
    unsigned kk = (s & 7u) + hi * 8u + (s >> 3) * 16u;
    float v = W1[(e * INDIM + kS * 32u + kk) * DIM + nT * 16u + n];
    pk.b[j] = (__bf16)v;
  }
  ((unsigned int*)W1b)[g] = pk.u;
}

__global__ void pack_w2(const float* __restrict__ W2, bf16_t* __restrict__ W2b) {
  unsigned g = blockIdx.x * 256u + threadIdx.x;       // total 4*8*32*256
  unsigned tile = g >> 8, d = g & 255u;
  unsigned lane = d >> 3, sp = d & 7u;
  unsigned hi = lane >> 4, n = lane & 15u;
  unsigned e = tile >> 8, oT = (tile >> 5) & 7u, kS = tile & 31u;
  BFPK pk;
#pragma unroll
  for (int j = 0; j < 2; ++j) {
    unsigned s  = sp * 2u + (unsigned)j;
    unsigned kk = (s & 7u) + hi * 8u + (s >> 3) * 16u;
    float v = W2[(e * DIM + kS * 32u + kk) * OUTD + oT * 16u + n];
    pk.b[j] = (__bf16)v;
  }
  ((unsigned int*)W2b)[g] = pk.u;
}

// ---------------- fused MoE main kernel ----------------
__global__ void __launch_bounds__(NTHR) moe_main(
    const int*   __restrict__ x,
    const float* __restrict__ emb0, const float* __restrict__ emb1,
    const float* __restrict__ b1,   const float* __restrict__ b2,
    const float* __restrict__ Wg,   const float* __restrict__ bg,
    const bf16_t* __restrict__ W1b, const bf16_t* __restrict__ W2b,
    float* __restrict__ out)
{
  extern __shared__ char smem[];
  bf16_t* sE = (bf16_t*)smem;                                        // [BM][EPITCH]
  bf16_t* sH = (bf16_t*)(smem + BM * EPITCH * 2);                    // [BM][HPITCH]
  float*  sG = (float*) (smem + BM * EPITCH * 2 + BM * HPITCH * 2);  // [BM][NEXP]

  const int t    = threadIdx.x;
  const int blk  = blockIdx.x;
  const int lane = t & 31;
  const int w    = t >> 5;          // wave 0..15
  const int nlo  = lane & 15;
  const int hi   = lane >> 4;
  const int mT   = w >> 3;          // M tile 0..1 (rows mT*16 .. mT*16+15)
  const int wg   = w & 7;           // N group (GEMM1) / OUT tile (GEMM2)

  // ---- stage 0: gather e tile (concat emb0/emb1 rows) -> LDS bf16 ----
  for (int idx = t; idx < BM * (INDIM / 4); idx += NTHR) {
    int m  = idx >> 9;                 // INDIM/4 = 512 float4 per row
    int c4 = (idx & 511) * 4;
    int row = blk * BM + m;
    const float* src = (c4 < DIM)
        ? (emb0 + (long)x[row * 2 + 0] * DIM + c4)
        : (emb1 + (long)x[row * 2 + 1] * DIM + (c4 - DIM));
    float4 f = *(const float4*)src;
    v4bf b;
    b[0] = (__bf16)f.x; b[1] = (__bf16)f.y; b[2] = (__bf16)f.z; b[3] = (__bf16)f.w;
    *(v4bf*)(sE + m * EPITCH + c4) = b;
  }
  if (t < BM * NEXP) sG[t] = bg[t & (NEXP - 1)];
  __syncthreads();

  // ---- stage 1: gating logits + softmax ----
  {
    int m = t & (BM - 1);
    int part = t >> 5;                 // 16 partials per row, 128 elems each
    float l0 = 0.f, l1 = 0.f, l2 = 0.f, l3 = 0.f;
    int i0 = part * (INDIM / 16);
    for (int i = i0; i < i0 + (INDIM / 16); ++i) {
      float ev = (float)sE[m * EPITCH + i];
      const float* wgp = Wg + i * NEXP;
      l0 += ev * wgp[0]; l1 += ev * wgp[1]; l2 += ev * wgp[2]; l3 += ev * wgp[3];
    }
    atomicAdd(&sG[m * NEXP + 0], l0); atomicAdd(&sG[m * NEXP + 1], l1);
    atomicAdd(&sG[m * NEXP + 2], l2); atomicAdd(&sG[m * NEXP + 3], l3);
  }
  __syncthreads();
  if (t < BM) {
    float g0 = sG[t*NEXP], g1 = sG[t*NEXP+1], g2 = sG[t*NEXP+2], g3 = sG[t*NEXP+3];
    float mx = fmaxf(fmaxf(g0, g1), fmaxf(g2, g3));
    g0 = __expf(g0 - mx); g1 = __expf(g1 - mx); g2 = __expf(g2 - mx); g3 = __expf(g3 - mx);
    float inv = 1.0f / (g0 + g1 + g2 + g3);
    sG[t*NEXP] = g0*inv; sG[t*NEXP+1] = g1*inv; sG[t*NEXP+2] = g2*inv; sG[t*NEXP+3] = g3*inv;
  }
  __syncthreads();

  v8f oacc = {};   // persistent: out tile (rows mT*16.., cols wg*16..wg*16+15)

  for (int e = 0; e < NEXP; ++e) {
    // ---- GEMM1: h = silu(e @ W1[e] + b1[e]) ; wave owns n-tiles wg*8..wg*8+7, rows mT ----
    v8f acc[8];
#pragma unroll
    for (int i = 0; i < 8; ++i) acc[i] = (v8f){0,0,0,0,0,0,0,0};

    for (int kS = 0; kS < 64; ++kS) {
      V16U a;
      const bf16_t* ap = sE + (mT * 16 + nlo) * EPITCH + kS * 32 + hi * 8;
      a.h[0] = *(const v8bf*)(ap);
      a.h[1] = *(const v8bf*)(ap + 16);
      const bf16_t* bp = W1b + (((e * 64 + wg * 8) * 64) + kS) * 512 + lane * 16;
#pragma unroll
      for (int tt = 0; tt < 8; ++tt) {
        v16bf bfrag = ldB(bp + tt * (64 * 512));
        acc[tt] = wmma_bf16(a.v, bfrag, acc[tt]);
      }
    }
    // bias + SiLU -> LDS h tile (bf16)
#pragma unroll
    for (int tt = 0; tt < 8; ++tt) {
      int ncol = (wg * 8 + tt) * 16 + nlo;
      float bias = b1[e * DIM + ncol];
#pragma unroll
      for (int v = 0; v < 8; ++v) {
        float xv = acc[tt][v] + bias;
        float s  = xv / (1.0f + __expf(-xv));
        sH[(mT * 16 + v + 8 * hi) * HPITCH + ncol] = (__bf16)s;
      }
    }
    __syncthreads();

    // ---- GEMM2: expert_out tile = h @ W2[e] ; wave owns OUT tile (mT, oT=wg) ----
    v8f o2 = {};
    for (int kS = 0; kS < 32; ++kS) {
      V16U a;
      const bf16_t* ap = sH + (mT * 16 + nlo) * HPITCH + kS * 32 + hi * 8;
      a.h[0] = *(const v8bf*)(ap);
      a.h[1] = *(const v8bf*)(ap + 16);
      v16bf bfrag = ldB(W2b + (((e * 8 + wg) * 32) + kS) * 512 + lane * 16);
      o2 = wmma_bf16(a.v, bfrag, o2);
    }
    float b2v = b2[e * OUTD + wg * 16 + nlo];
#pragma unroll
    for (int v = 0; v < 8; ++v) {
      float g = sG[(mT * 16 + v + 8 * hi) * NEXP + e];
      oacc[v] += g * (o2[v] + b2v);
    }
    __syncthreads();   // sH reused next expert
  }

  // ---- write out [B, 128] f32 ----
#pragma unroll
  for (int v = 0; v < 8; ++v) {
    out[(blk * BM + mT * 16 + v + 8 * hi) * OUTD + wg * 16 + nlo] = oacc[v];
  }
}

// ---------------- launch ----------------
extern "C" void kernel_launch(void* const* d_in, const int* in_sizes, int n_in,
                              void* d_out, int out_size, void* d_ws, size_t ws_size,
                              hipStream_t stream) {
  const int*   x    = (const int*)d_in[0];
  const float* emb0 = (const float*)d_in[1];
  const float* emb1 = (const float*)d_in[2];
  const float* W1   = (const float*)d_in[3];
  const float* b1   = (const float*)d_in[4];
  const float* W2   = (const float*)d_in[5];
  const float* b2   = (const float*)d_in[6];
  const float* Wg   = (const float*)d_in[7];
  const float* bg   = (const float*)d_in[8];

  bf16_t* W1b = (bf16_t*)d_ws;                                   // 16,777,216 B
  bf16_t* W2b = (bf16_t*)((char*)d_ws + (size_t)16777216);       //  1,048,576 B

  pack_w1<<<16384, 256, 0, stream>>>(W1, W1b);
  pack_w2<<<1024,  256, 0, stream>>>(W2, W2b);

  const size_t smem = (size_t)BM * EPITCH * 2 + (size_t)BM * HPITCH * 2
                    + (size_t)BM * NEXP * 4;   // 198,144 B < 320 KB WGP LDS
  moe_main<<<B_TOT / BM, NTHR, smem, stream>>>(x, emb0, emb1, b1, b2, Wg, bg,
                                               W1b, W2b, (float*)d_out);
}
